// MultiHeadAttention_59700045415070
// MI455X (gfx1250) — matrix operand order repaired and checked by reading, not compiled, not run on hardware
//
#include <hip/hip_runtime.h>

// MI455X / gfx1250: wave32, WMMA 16x16x32 bf16 (f32 accumulate),
// async-to-LDS staging, split wait counters.

typedef __attribute__((ext_vector_type(16))) __bf16 v16bf;
typedef __attribute__((ext_vector_type(8)))  float  v8f;

union FragA { v16bf bf; unsigned int u[8]; };

__device__ __forceinline__ unsigned short f2bf(float x) {
    unsigned int u = __float_as_uint(x);
    u += 0x7FFFu + ((u >> 16) & 1u);   // round-to-nearest-even
    return (unsigned short)(u >> 16);
}
__device__ __forceinline__ unsigned int pack2(unsigned short lo, unsigned short hi) {
    return (unsigned int)lo | ((unsigned int)hi << 16);
}

// GLOBAL_LOAD_ASYNC_TO_LDS_B128 (GV mode): VDST = per-lane LDS byte offset,
// VADDR = 64-bit global address. Tracked by ASYNCcnt.
__device__ __forceinline__ void async_load_b128(unsigned ldsOff, const void* gptr) {
    asm volatile("global_load_async_to_lds_b128 %0, %1, off"
                 :: "v"(ldsOff), "v"(gptr) : "memory");
}
__device__ __forceinline__ void wait_async0() {
    asm volatile("s_wait_asynccnt 0" ::: "memory");
}
__device__ __forceinline__ void wait_ds0() {
    asm volatile("s_wait_dscnt 0" ::: "memory");
}

// ---------------------------------------------------------------------------
// GEMM: Y = X @ W^T + bias.  X:[M,1024], W:[1024,1024] f32 row-major.
// MODE 0: X f32, output bf16 re-laid-out as [B,H,S,64] (head split).
// MODE 1: X bf16 [M,1024], output f32 flat [M,1024].
// Block: 256 thr = 8 waves; tile 128x128; wave tile 32x64 (2x4 WMMA frags).
// ---------------------------------------------------------------------------
template<int MODE>
__global__ __launch_bounds__(256)
void gemm_wmma(const void* __restrict__ xin, const float* __restrict__ W,
               const float* __restrict__ bias, void* __restrict__ out)
{
    __shared__ __align__(16) unsigned short Xs[128 * 64];
    __shared__ __align__(16) unsigned short Ws[128 * 64];

    const int m0   = blockIdx.x * 128;
    const int n0   = blockIdx.y * 128;
    const int t    = threadIdx.x;
    const int lane = t & 31;
    const int wave = t >> 5;
    const int wm   = wave & 3;       // 4 waves along M  -> rows wm*32
    const int wn   = wave >> 2;      // 2 waves along N  -> cols wn*64
    const int lh   = lane >> 4;      // lane half (ISA A/B K-split)
    const int ll   = lane & 15;

    // A-fragment packed-pair pattern (uint index): K = lh*8 + {0,2,4,6,16,18,20,22}
    const int qpat[8] = {0, 1, 2, 3, 8, 9, 10, 11};

    v8f acc[2][4];
    #pragma unroll
    for (int a = 0; a < 2; ++a)
        #pragma unroll
        for (int b = 0; b < 4; ++b)
            #pragma unroll
            for (int e = 0; e < 8; ++e) acc[a][b][e] = 0.0f;

    for (int kb = 0; kb < 1024; kb += 64) {
        __syncthreads();
        // ---- stage 128x64 tiles of X and W into LDS as bf16 ----
        #pragma unroll
        for (int j = 0; j < 8; ++j) {
            int idx4 = t + j * 256;          // float4-granule index in tile
            int r = idx4 >> 4;
            int c = (idx4 & 15) << 2;
            unsigned int lo, hi;
            if (MODE == 0) {
                const float* xg = (const float*)xin;
                float4 vv = *(const float4*)(xg + (size_t)(m0 + r) * 1024 + kb + c);
                lo = pack2(f2bf(vv.x), f2bf(vv.y));
                hi = pack2(f2bf(vv.z), f2bf(vv.w));
            } else {
                const unsigned short* xg = (const unsigned short*)xin;
                uint2 vv = *(const uint2*)(xg + (size_t)(m0 + r) * 1024 + kb + c);
                lo = vv.x; hi = vv.y;
            }
            *(uint2*)&Xs[r * 64 + c] = make_uint2(lo, hi);

            float4 ww = *(const float4*)(W + (size_t)(n0 + r) * 1024 + kb + c);
            *(uint2*)&Ws[r * 64 + c] =
                make_uint2(pack2(f2bf(ww.x), f2bf(ww.y)), pack2(f2bf(ww.z), f2bf(ww.w)));
        }
        // prefetch next K-tile (global_prefetch_b8)
        if (kb + 64 < 1024) {
            int r = t & 127;
            if (MODE == 0)
                __builtin_prefetch((const float*)xin + (size_t)(m0 + r) * 1024 + kb + 64, 0, 1);
            else
                __builtin_prefetch((const unsigned short*)xin + (size_t)(m0 + r) * 1024 + kb + 64, 0, 1);
            __builtin_prefetch(W + (size_t)(n0 + r) * 1024 + kb + 64, 0, 1);
        }
        __syncthreads();

        const unsigned int* Xs32 = (const unsigned int*)Xs;
        const unsigned int* Ws32 = (const unsigned int*)Ws;
        #pragma unroll
        for (int ks = 0; ks < 2; ++ks) {
            FragA a[2], b[4];
            #pragma unroll
            for (int tm = 0; tm < 2; ++tm) {
                int row  = wm * 32 + tm * 16 + ll;
                int base = row * 32 + ks * 16 + lh * 4;
                #pragma unroll
                for (int i = 0; i < 8; ++i) a[tm].u[i] = Xs32[base + qpat[i]];
            }
            #pragma unroll
            for (int tn = 0; tn < 4; ++tn) {
                int nr   = wn * 64 + tn * 16 + ll;
                int base = nr * 32 + ks * 16 + lh * 4;
                #pragma unroll
                for (int i = 0; i < 8; ++i) b[tn].u[i] = Ws32[base + qpat[i]];
            }
            #pragma unroll
            for (int tm = 0; tm < 2; ++tm)
                #pragma unroll
                for (int tn = 0; tn < 4; ++tn)
                    acc[tm][tn] = __builtin_amdgcn_wmma_f32_16x16x32_bf16(
                        false, a[tm].bf, false, b[tn].bf,
                        (short)0, acc[tm][tn], false, false);
        }
    }

    // ---- epilogue: bias + store ----
    #pragma unroll
    for (int tm = 0; tm < 2; ++tm) {
        #pragma unroll
        for (int tn = 0; tn < 4; ++tn) {
            int n = n0 + wn * 64 + tn * 16 + ll;
            float bvv = bias[n];
            #pragma unroll
            for (int i = 0; i < 8; ++i) {
                int m = m0 + wm * 32 + tm * 16 + i + 8 * lh;
                float y = acc[tm][tn][i] + bvv;
                if (MODE == 0) {
                    unsigned short* o = (unsigned short*)out;
                    int bidx = m >> 11, s = m & 2047;   // S = 2048
                    int h = n >> 6, d = n & 63;
                    o[((((size_t)bidx * 16 + h) * 2048) + s) * 64 + d] = f2bf(y);
                } else {
                    float* o = (float*)out;
                    o[(size_t)m * 1024 + n] = y;
                }
            }
        }
    }
}

// ---------------------------------------------------------------------------
// Causal flash attention. Qh/Kh/Vh bf16 [B,H,S,64]. Out bf16 [B,S,1024].
// 256 thr = 8 waves, 128 queries/block. K/V 32-key blocks staged once per
// workgroup in LDS via GLOBAL_LOAD_ASYNC_TO_LDS_B128 (ASYNCcnt), uniform
// kv-loop with wave-uniform compute guards (causal skip).
// ---------------------------------------------------------------------------
__global__ __launch_bounds__(256)
void attn_wmma(const unsigned short* __restrict__ Qh,
               const unsigned short* __restrict__ Kh,
               const unsigned short* __restrict__ Vh,
               unsigned short* __restrict__ Out)
{
    __shared__ __align__(16) unsigned short Ks[32 * 64];     // key block
    __shared__ __align__(16) unsigned short Vs[32 * 64];     // value block
    __shared__ __align__(16) unsigned short Ps[8][16 * 32];  // P staging per wave

    const int t    = threadIdx.x;
    const int lane = t & 31;
    const int wave = t >> 5;
    const int lh   = lane >> 4;
    const int ll   = lane & 15;

    const int blk  = blockIdx.x;
    const int qblk = blk & 15;
    const int h    = (blk >> 4) & 15;
    const int b    = blk >> 8;

    const size_t headoff = (((size_t)b * 16 + h) * 2048) * 64;
    const unsigned short* qh = Qh + headoff;
    const unsigned short* kh = Kh + headoff;
    const unsigned short* vh = Vh + headoff;

    const int q0 = qblk * 128;           // block's first query row
    const int qw = q0 + wave * 16;       // this wave's first query row
    const int qpat[8] = {0, 1, 2, 3, 8, 9, 10, 11};

    // LDS byte offsets (generic ptr low 32 bits == LDS offset, ISA 10.2)
    const unsigned ksBase = (unsigned)(size_t)(void*)Ks;
    const unsigned vsBase = (unsigned)(size_t)(void*)Vs;
    const int srow = t >> 3;             // staging row 0..31
    const int sch  = t & 7;              // 16B chunk within 128B row

    // Q A-fragments: d 0..31 and d 32..63
    FragA qf[2];
    {
        const unsigned int* q32 = (const unsigned int*)(qh + (size_t)(qw + ll) * 64);
        #pragma unroll
        for (int i = 0; i < 8; ++i) {
            qf[0].u[i] = q32[lh * 4 + qpat[i]];
            qf[1].u[i] = q32[16 + lh * 4 + qpat[i]];
        }
    }

    float mrow[8], lrow[8];
    #pragma unroll
    for (int i = 0; i < 8; ++i) { mrow[i] = -3.0e38f; lrow[i] = 0.0f; }
    v8f acc[4];
    #pragma unroll
    for (int d = 0; d < 4; ++d)
        #pragma unroll
        for (int e = 0; e < 8; ++e) acc[d][e] = 0.0f;

    const float invScale = 0.125f;   // 1/sqrt(64)

    for (int kb = 0; kb < q0 + 128; kb += 32) {
        // ---- async-stage K and V 32x64 bf16 blocks into LDS ----
        async_load_b128(ksBase + srow * 128 + sch * 16,
                        (const char*)(kh + (size_t)(kb + srow) * 64) + sch * 16);
        async_load_b128(vsBase + srow * 128 + sch * 16,
                        (const char*)(vh + (size_t)(kb + srow) * 64) + sch * 16);
        wait_async0();
        __syncthreads();

        if (kb < qw + 16) {   // wave-uniform causal guard
            // ---- S = Q @ K^T for 32 keys (2 N-tiles x 2 K-steps) ----
            const unsigned int* Ks32 = (const unsigned int*)Ks;
            v8f s[2];
            #pragma unroll
            for (int tn = 0; tn < 2; ++tn)
                #pragma unroll
                for (int e = 0; e < 8; ++e) s[tn][e] = 0.0f;
            #pragma unroll
            for (int tn = 0; tn < 2; ++tn) {
                int rbase = (tn * 16 + ll) * 32;
                FragA bk0, bk1;
                #pragma unroll
                for (int i = 0; i < 8; ++i) {
                    bk0.u[i] = Ks32[rbase + lh * 4 + qpat[i]];
                    bk1.u[i] = Ks32[rbase + 16 + lh * 4 + qpat[i]];
                }
                s[tn] = __builtin_amdgcn_wmma_f32_16x16x32_bf16(
                    false, qf[0].bf, false, bk0.bf, (short)0, s[tn], false, false);
                s[tn] = __builtin_amdgcn_wmma_f32_16x16x32_bf16(
                    false, qf[1].bf, false, bk1.bf, (short)0, s[tn], false, false);
            }

            // ---- scale, causal mask, online softmax stats ----
            float p[2][8], alpha[8];
            #pragma unroll
            for (int i = 0; i < 8; ++i) {
                int qrow = qw + i + 8 * lh;
                float v0 = s[0][i] * invScale;
                float v1 = s[1][i] * invScale;
                if (kb + ll > qrow)      v0 = -1.0e9f;
                if (kb + 16 + ll > qrow) v1 = -1.0e9f;
                float mx = fmaxf(v0, v1);
                #pragma unroll
                for (int off = 1; off < 16; off <<= 1)
                    mx = fmaxf(mx, __shfl_xor(mx, off, 32));
                float mn = fmaxf(mrow[i], mx);
                float a  = __expf(mrow[i] - mn);
                float p0 = __expf(v0 - mn);
                float p1 = __expf(v1 - mn);
                float rs = p0 + p1;
                #pragma unroll
                for (int off = 1; off < 16; off <<= 1)
                    rs += __shfl_xor(rs, off, 32);
                lrow[i] = lrow[i] * a + rs;
                mrow[i] = mn;
                alpha[i] = a;
                p[0][i] = p0; p[1][i] = p1;
            }
            #pragma unroll
            for (int d = 0; d < 4; ++d)
                #pragma unroll
                for (int i = 0; i < 8; ++i) acc[d][i] *= alpha[i];

            // ---- P: C-layout -> LDS -> A-layout (per-wave region) ----
            #pragma unroll
            for (int tn = 0; tn < 2; ++tn)
                #pragma unroll
                for (int i = 0; i < 8; ++i)
                    Ps[wave][(i + 8 * lh) * 32 + tn * 16 + ll] = f2bf(p[tn][i]);
            wait_ds0();

            FragA pf;
            {
                const unsigned int* p32 = (const unsigned int*)&Ps[wave][ll * 32];
                #pragma unroll
                for (int i = 0; i < 8; ++i) pf.u[i] = p32[lh * 8 + i];
            }

            // ---- O += P @ V (4 d-tiles, K = 32 keys) ----
            #pragma unroll
            for (int d = 0; d < 4; ++d) {
                FragA bv;
                int col = d * 16 + ll;
                #pragma unroll
                for (int i = 0; i < 8; ++i) {
                    int krel = lh * 16 + 2 * i;
                    unsigned short lo = Vs[krel * 64 + col];
                    unsigned short hi = Vs[(krel + 1) * 64 + col];
                    bv.u[i] = pack2(lo, hi);
                }
                acc[d] = __builtin_amdgcn_wmma_f32_16x16x32_bf16(
                    false, pf.bf, false, bv.bf, (short)0, acc[d], false, false);
            }
        }
        __syncthreads();   // protect K/V re-stage
    }

    // ---- normalize and store bf16 [B,S,1024] ----
    unsigned short* outp = Out + ((size_t)b * 2048) * 1024;
    #pragma unroll
    for (int d = 0; d < 4; ++d) {
        int col = h * 64 + d * 16 + ll;
        #pragma unroll
        for (int i = 0; i < 8; ++i) {
            int q = qw + i + 8 * lh;
            outp[(size_t)q * 1024 + col] = f2bf(acc[d][i] / lrow[i]);
        }
    }
}

// ---------------------------------------------------------------------------
extern "C" void kernel_launch(void* const* d_in, const int* in_sizes, int n_in,
                              void* d_out, int out_size, void* d_ws, size_t ws_size,
                              hipStream_t stream)
{
    (void)in_sizes; (void)n_in; (void)out_size; (void)ws_size;
    const float* q  = (const float*)d_in[0];
    const float* k  = (const float*)d_in[1];
    const float* v  = (const float*)d_in[2];
    // d_in[3] = causal mask; implemented analytically (k <= q)
    const float* Wq = (const float*)d_in[4];
    const float* bq = (const float*)d_in[5];
    const float* Wk = (const float*)d_in[6];
    const float* bk = (const float*)d_in[7];
    const float* Wv = (const float*)d_in[8];
    const float* bv = (const float*)d_in[9];
    const float* Wo = (const float*)d_in[10];
    const float* bo = (const float*)d_in[11];

    const size_t headElems = (size_t)4 * 16 * 2048 * 64;   // 8,388,608
    unsigned short* Qh = (unsigned short*)d_ws;
    unsigned short* Kh = Qh + headElems;
    unsigned short* Vh = Kh + headElems;
    unsigned short* AO = Vh + headElems;                   // [B,S,1024] bf16

    dim3 grid(64, 8), block(256);
    gemm_wmma<0><<<grid, block, 0, stream>>>(q, Wq, bq, Qh);
    gemm_wmma<0><<<grid, block, 0, stream>>>(k, Wk, bk, Kh);
    gemm_wmma<0><<<grid, block, 0, stream>>>(v, Wv, bv, Vh);
    attn_wmma<<<dim3(1024), block, 0, stream>>>(Qh, Kh, Vh, AO);
    gemm_wmma<1><<<grid, block, 0, stream>>>(AO, Wo, bo, d_out);
}
